// GraphResConv_12824772346522
// MI455X (gfx1250) — matrix operand order repaired and checked by reading, not compiled, not run on hardware
//
#include <hip/hip_runtime.h>

#define D 512
#define EPS_BN 1e-5f
#define BM 128
#define BNT 64
#define BK 32
#define LDA 40   // halves per LDS row (80B: 16B-aligned, conflict-free frag reads)
#define LDB 40

typedef __attribute__((ext_vector_type(16))) _Float16 v16h;
typedef __attribute__((ext_vector_type(8)))  _Float16 v8h;
typedef __attribute__((ext_vector_type(4)))  _Float16 v4h;
typedef __attribute__((ext_vector_type(8)))  float    v8f;

__global__ __launch_bounds__(256) void copy_f32(const float* __restrict__ in,
                                                float* __restrict__ out, int n4) {
  int i = blockIdx.x * blockDim.x + threadIdx.x;
  if (i < n4) ((float4*)out)[i] = ((const float4*)in)[i];
}

__global__ __launch_bounds__(256) void zero_f32(float* __restrict__ p, int n) {
  int i = blockIdx.x * blockDim.x + threadIdx.x;
  if (i < n) p[i] = 0.f;
}

// acc[dst] += src[srcRow]; one thread per (edge, 4-float chunk). Accumulator fits in L2.
__global__ __launch_bounds__(256) void scatter_add(const float* __restrict__ src,
                                                   const long long* __restrict__ ei,
                                                   float* __restrict__ acc, int E) {
  int tid = blockIdx.x * blockDim.x + threadIdx.x;
  int e = tid >> 7;              // D/4 = 128 chunks per edge
  int c = (tid & 127) << 2;
  if (e >= E) return;
  int s = (int)ei[e];
  int d = (int)ei[(size_t)E + e];
  float4 v = *(const float4*)(src + (size_t)s * D + c);
  float* p = acc + (size_t)d * D + c;
  atomicAdd(p + 0, v.x); atomicAdd(p + 1, v.y);
  atomicAdd(p + 2, v.z); atomicAdd(p + 3, v.w);
}

// wt[n][k] = (f16) w[k][n]  (transpose so B fragments read contiguous K-pairs)
__global__ __launch_bounds__(256) void wcvt_t(const float* __restrict__ w,
                                              _Float16* __restrict__ wt) {
  int idx = blockIdx.x * blockDim.x + threadIdx.x;   // 0..D*D-1, idx = k*D + n
  int k = idx >> 9, n = idx & (D - 1);
  wt[(size_t)n * D + k] = (_Float16)w[idx];
}

// out = act(A @ W + bias). A read as f32 (converted during staging) or f16.
// Output written as f32 (for BN) or f16 (feeding the next GEMM directly).
// TAIL=false instantiation is guard-free straight-line code for full 128-row tiles.
template <bool TAIL>
__global__ __launch_bounds__(256)
void wmma_gemm(const float* __restrict__ A32, const _Float16* __restrict__ A16,
               const _Float16* __restrict__ WT, const float* __restrict__ bias,
               float* __restrict__ out32, _Float16* __restrict__ out16,
               int Mrows, int doRelu, int rowBlockOff) {
  __shared__ _Float16 As[BM * LDA];
  __shared__ _Float16 Bs[BNT * LDB];

  int tid  = threadIdx.x;
  int lane = tid & 31, wave = tid >> 5;
  int wm = wave & 3, wn = wave >> 2;          // 4x2 wave grid, 32x32 per wave
  int rowBase = (blockIdx.x + rowBlockOff) * BM;
  int colBase = blockIdx.y * BNT;
  int g  = lane >> 4;                          // half-wave group
  int lm = lane & 15;

  v8f acc[2][2] = {};

  for (int kc = 0; kc < D; kc += BK) {
    // Stage B tile asynchronously: weights, always in-bounds; per-lane LDS dest.
    {
      int r = tid >> 2, ch = (tid & 3) * 8;
      const _Float16* gp = WT + (size_t)(colBase + r) * D + kc + ch;
      unsigned lp = (unsigned)(size_t)&Bs[r * LDB + ch];
      asm volatile("global_load_async_to_lds_b128 %0, %1, off"
                   :: "v"(lp), "v"(gp) : "memory");
    }

    // Stage A tile: 128 rows x 32 halves, converting f32->f16 on the fly.
    if (A32) {
#pragma unroll
      for (int i = 0; i < 4; ++i) {
        int idx = tid + i * 256;
        int r = idx >> 3, ch = (idx & 7) * 4;
        int gr = rowBase + r;
        float4 v = {0.f, 0.f, 0.f, 0.f};
        if (!TAIL || gr < Mrows) {
          const float* ap = A32 + (size_t)gr * D + kc + ch;
          v = *(const float4*)ap;
          __builtin_prefetch(ap + BK, 0, 3);   // next K chunk; speculative OOB is dropped
        }
        v4h h;
        h[0] = (_Float16)v.x; h[1] = (_Float16)v.y;
        h[2] = (_Float16)v.z; h[3] = (_Float16)v.w;
        *(v4h*)&As[r * LDA + ch] = h;
      }
    } else {
#pragma unroll
      for (int i = 0; i < 2; ++i) {
        int idx = tid + i * 256;
        int r = idx >> 2, ch = (idx & 3) * 8;
        int gr = rowBase + r;
        v8h hv = {};
        if (!TAIL || gr < Mrows) {
          const _Float16* ap = A16 + (size_t)gr * D + kc + ch;
          hv = *(const v8h*)ap;
          __builtin_prefetch(ap + BK, 0, 3);
        }
        *(v8h*)&As[r * LDA + ch] = hv;
      }
    }

    asm volatile("s_wait_asynccnt 0x0" ::: "memory");
    __syncthreads();

    // Build fragments per ISA 7.12.2 16-bit layouts (pair loads as dwords)
    union F { v16h v; unsigned int u[8]; };
    F a0, a1, b0, b1;
#pragma unroll
    for (int v = 0; v < 8; ++v) {
      int ka = ((v & 4) ? 16 : 0) + g * 8 + (v & 3) * 2;   // A 16x32 K index
      a0.u[v] = *(const unsigned int*)&As[(wm * 32 + lm) * LDA + ka];
      a1.u[v] = *(const unsigned int*)&As[(wm * 32 + 16 + lm) * LDA + ka];
      int kb = ((v & 4) ? 16 : 0) + g * 8 + (v & 3) * 2;   // B 32x16 K index
      b0.u[v] = *(const unsigned int*)&Bs[(wn * 32 + lm) * LDB + kb];
      b1.u[v] = *(const unsigned int*)&Bs[(wn * 32 + 16 + lm) * LDB + kb];
    }
    acc[0][0] = __builtin_amdgcn_wmma_f32_16x16x32_f16(false, a0.v, false, b0.v,
                                                       (short)0, acc[0][0], false, false);
    acc[0][1] = __builtin_amdgcn_wmma_f32_16x16x32_f16(false, a0.v, false, b1.v,
                                                       (short)0, acc[0][1], false, false);
    acc[1][0] = __builtin_amdgcn_wmma_f32_16x16x32_f16(false, a1.v, false, b0.v,
                                                       (short)0, acc[1][0], false, false);
    acc[1][1] = __builtin_amdgcn_wmma_f32_16x16x32_f16(false, a1.v, false, b1.v,
                                                       (short)0, acc[1][1], false, false);
    __syncthreads();
  }

  // Epilogue: C/D layout M = g*8 + v, N = lane%16; fused bias + ReLU (+ f16 cvt)
#pragma unroll
  for (int sm = 0; sm < 2; ++sm) {
#pragma unroll
    for (int sn = 0; sn < 2; ++sn) {
      int col = colBase + wn * 32 + sn * 16 + lm;
      float bv = bias[col];
      int rb = rowBase + wm * 32 + sm * 16 + g * 8;
#pragma unroll
      for (int v = 0; v < 8; ++v) {
        int r = rb + v;
        if (!TAIL || r < Mrows) {
          float val = acc[sm][sn][v] + bv;
          if (doRelu) val = fmaxf(val, 0.f);
          if (out16) out16[(size_t)r * D + col] = (_Float16)val;
          else       out32[(size_t)r * D + col] = val;
        }
      }
    }
  }
}

// Column sums / sumsq partials -> global atomics. Thread t owns cols t and t+256.
__global__ __launch_bounds__(256)
void bn_stats(const float* __restrict__ h, float* __restrict__ stats, int rows, int rpb) {
  int c = threadIdx.x;
  int r0 = blockIdx.x * rpb;
  int r1 = r0 + rpb; if (r1 > rows) r1 = rows;
  float s0 = 0.f, q0 = 0.f, s1 = 0.f, q1 = 0.f;
  for (int r = r0; r < r1; ++r) {
    float a = h[(size_t)r * D + c];
    float b = h[(size_t)r * D + c + 256];
    s0 += a; q0 += a * a; s1 += b; q1 += b * b;
  }
  atomicAdd(&stats[c], s0);
  atomicAdd(&stats[c + 256], s1);
  atomicAdd(&stats[D + c], q0);
  atomicAdd(&stats[D + c + 256], q1);
}

// out = relu((h - mu) * rsqrt(var+eps) * gamma + beta) [+ identity]; optional dual write
__global__ __launch_bounds__(256)
void bn_apply(const float* __restrict__ h, const float* __restrict__ stats,
              const float* __restrict__ gamma, const float* __restrict__ beta,
              const float* __restrict__ identity, float* __restrict__ out,
              float* __restrict__ out2, int n4, float invN) {
  int i = blockIdx.x * blockDim.x + threadIdx.x;
  if (i >= n4) return;
  int base = i * 4;
  int c = base & (D - 1);
  float4 v = ((const float4*)h)[i];
  float o[4] = {v.x, v.y, v.z, v.w};
  float r[4];
#pragma unroll
  for (int j = 0; j < 4; ++j) {
    int col = c + j;
    float mu  = stats[col] * invN;
    float var = stats[D + col] * invN - mu * mu;
    float val = (o[j] - mu) * rsqrtf(var + EPS_BN) * gamma[col] + beta[col];
    r[j] = fmaxf(val, 0.f);
  }
  float4 res = {r[0], r[1], r[2], r[3]};
  if (identity) {
    float4 idv = ((const float4*)identity)[i];
    res.x += idv.x; res.y += idv.y; res.z += idv.z; res.w += idv.w;
  }
  ((float4*)out)[i] = res;
  if (out2) ((float4*)out2)[i] = res;
}

// Helper: launch full-tile blocks guard-free, tail block separately.
static inline void launch_gemm(const float* A32, const _Float16* A16,
                               const _Float16* WT, const float* bias,
                               float* out32, _Float16* out16,
                               int Nn, int doRelu, hipStream_t stream) {
  int fullBlocks = Nn / BM;
  int tailRows   = Nn - fullBlocks * BM;
  dim3 blk(256);
  if (fullBlocks > 0) {
    dim3 grid(fullBlocks, D / BNT);
    wmma_gemm<false><<<grid, blk, 0, stream>>>(A32, A16, WT, bias, out32, out16,
                                               Nn, doRelu, 0);
  }
  if (tailRows > 0) {
    dim3 grid(1, D / BNT);
    wmma_gemm<true><<<grid, blk, 0, stream>>>(A32, A16, WT, bias, out32, out16,
                                              Nn, doRelu, fullBlocks);
  }
}

extern "C" void kernel_launch(void* const* d_in, const int* in_sizes, int n_in,
                              void* d_out, int out_size, void* d_ws, size_t ws_size,
                              hipStream_t stream) {
  const float*     x   = (const float*)d_in[0];
  const long long* ei  = (const long long*)d_in[1];
  const float* w1a = (const float*)d_in[2];  const float* b1a = (const float*)d_in[3];
  const float* w1b = (const float*)d_in[4];  const float* b1b = (const float*)d_in[5];
  const float* g1  = (const float*)d_in[6];  const float* be1 = (const float*)d_in[7];
  const float* w2a = (const float*)d_in[8];  const float* b2a = (const float*)d_in[9];
  const float* w2b = (const float*)d_in[10]; const float* b2b = (const float*)d_in[11];
  const float* g2  = (const float*)d_in[12]; const float* be2 = (const float*)d_in[13];

  int Nn = in_sizes[0] / D;
  int E  = in_sizes[1] / 2;
  size_t ND = (size_t)Nn * D;

  float*    bufA = (float*)d_ws;
  float*    bufB = bufA + ND;
  _Float16* a16  = (_Float16*)(bufB + ND);
  _Float16* w16  = a16 + ND;
  float*    stats = (float*)(w16 + (size_t)4 * D * D);

  _Float16* wt1a = w16;
  _Float16* wt1b = w16 + (size_t)D * D;
  _Float16* wt2a = w16 + (size_t)2 * D * D;
  _Float16* wt2b = w16 + (size_t)3 * D * D;

  dim3 blk(256);
  int n4 = (int)(ND / 4);
  int elwBlocks  = (n4 + 255) / 256;
  int scatBlocks = (int)(((size_t)E * (D / 4) + 255) / 256);
  int wBlocks    = (D * D + 255) / 256;
  int rpb = (Nn + 255) / 256;
  float invN = 1.f / (float)Nn;

  // One-time (per launch) f16 transposed weights
  wcvt_t<<<wBlocks, blk, 0, stream>>>(w1a, wt1a);
  wcvt_t<<<wBlocks, blk, 0, stream>>>(w1b, wt1b);
  wcvt_t<<<wBlocks, blk, 0, stream>>>(w2a, wt2a);
  wcvt_t<<<wBlocks, blk, 0, stream>>>(w2b, wt2b);

  // ---- GIN layer 1 ----
  copy_f32   <<<elwBlocks, blk, 0, stream>>>(x, bufA, n4);
  scatter_add<<<scatBlocks, blk, 0, stream>>>(x, ei, bufA, E);
  launch_gemm(bufA, nullptr, wt1a, b1a, nullptr, a16, Nn, 1, stream);
  launch_gemm(nullptr, a16, wt1b, b1b, bufB, nullptr, Nn, 0, stream);
  zero_f32   <<<(2 * D + 255) / 256, blk, 0, stream>>>(stats, 2 * D);
  bn_stats   <<<256, blk, 0, stream>>>(bufB, stats, Nn, rpb);
  // h1n -> bufA (scatter source) and bufB (accumulator init), fused copy
  bn_apply   <<<elwBlocks, blk, 0, stream>>>(bufB, stats, g1, be1, nullptr,
                                             bufA, bufB, n4, invN);

  // ---- GIN layer 2 ----
  scatter_add<<<scatBlocks, blk, 0, stream>>>(bufA, ei, bufB, E);
  launch_gemm(bufB, nullptr, wt2a, b2a, nullptr, a16, Nn, 1, stream);
  launch_gemm(nullptr, a16, wt2b, b2b, bufA, nullptr, Nn, 0, stream);
  zero_f32   <<<(2 * D + 255) / 256, blk, 0, stream>>>(stats, 2 * D);
  bn_stats   <<<256, blk, 0, stream>>>(bufA, stats, Nn, rpb);
  bn_apply   <<<elwBlocks, blk, 0, stream>>>(bufA, stats, g2, be2, x,
                                             (float*)d_out, nullptr, n4, invN);
}
